// GraphNetBlock_88734024336034
// MI455X (gfx1250) — matrix operand order repaired and checked by reading, not compiled, not run on hardware
//
#include <hip/hip_runtime.h>
#include <hip/hip_bf16.h>

// ---------------------------------------------------------------------------
// GraphNet block for MI455X (gfx1250, wave32, WMMA).
// Strategy: bf16 WMMA (v_wmma_f32_16x16x32_bf16) for all GEMMs, in-register
// LayerNorm via __shfl_xor within 16-lane halves, LDS-staged A tiles and
// pre-swizzled weight fragments, segment-sum via global_atomic_add_f32.
// ---------------------------------------------------------------------------

#define N_NODES   50000
#define N_EDGES   800000
#define HID       128

typedef __attribute__((ext_vector_type(8)))  float        v8f;
typedef __attribute__((ext_vector_type(4)))  unsigned int u32x4;
typedef __attribute__((ext_vector_type(16))) __bf16       v16bf;

union BFfrag { struct { u32x4 lo, hi; } s; v16bf v; };

__device__ __forceinline__ unsigned short f2bf(float f) {
    unsigned int u = __builtin_bit_cast(unsigned int, f);
    unsigned int r = u + 0x7FFFu + ((u >> 16) & 1u);   // round-to-nearest-even
    return (unsigned short)(r >> 16);
}

// --------------------------------------------------------------------------
// Weight prep: f32 row-major W[K][128] -> bf16 WMMA B-fragments.
// Per (ntile t, kchunk c): 32 lanes x 8 dwords. Lane layout (ISA 7.12.2):
//   col = t*16 + lane%16 ; dword j<4: K = c*32 + (lane/16)*8 + 2j,+1 ; j>=4: K = c*32 + 16 + (lane/16)*8 + 2(j-4),+1
// --------------------------------------------------------------------------
__global__ void prep_w_kernel(const float* __restrict__ W, int K, int kchunks,
                              unsigned int* __restrict__ dst) {
    int b    = blockIdx.x;          // b = t*kchunks + c
    int tid  = threadIdx.x;
    int lane = tid & 31;
    int j    = tid >> 5;            // 0..7
    int c    = b % kchunks;
    int t    = b / kchunks;
    int col  = t * 16 + (lane & 15);
    int k0   = c * 32 + ((lane >> 4) << 3) + 2 * j + ((j >= 4) ? 8 : 0);
    float a  = (k0     < K) ? W[(size_t)k0       * HID + col] : 0.f;
    float bb = (k0 + 1 < K) ? W[(size_t)(k0 + 1) * HID + col] : 0.f;
    unsigned int p = (unsigned int)f2bf(a) | ((unsigned int)f2bf(bb) << 16);
    dst[((size_t)b * 32 + lane) * 8 + j] = p;
}

__global__ void zero_agg_kernel(float* __restrict__ agg) {
    int i = (blockIdx.x * blockDim.x + threadIdx.x) * 4;
    *(float4*)(agg + i) = make_float4(0.f, 0.f, 0.f, 0.f);   // 6250*256*4 = 6.4M
}

// -------------------------- shared device pieces --------------------------
__device__ __forceinline__ void copy_w(const unsigned int* __restrict__ src,
                                       unsigned int* __restrict__ dst,
                                       int n_u32, int tid) {
    for (int i = tid * 4; i < n_u32; i += 256 * 4)
        *(u32x4*)(dst + i) = *(const u32x4*)(src + i);
}

// One layer GEMM: A(16 x 32k, bf16, LDS) x B(32k x 128, fragments in LDS)
// accumulated into 8 tiles of 16x16 f32 per wave.
template <int KPAD>
__device__ __forceinline__ void wmma_gemm(const unsigned short* __restrict__ lds_a,
                                          const unsigned int*   __restrict__ lds_w,
                                          int kchunks, int wave, int lane, v8f* acc) {
    const int arow  = wave * 16 + (lane & 15);
    const int koff0 = (lane < 16) ? 0 : 8;
    for (int c = 0; c < kchunks; ++c) {
        BFfrag A;
        const unsigned short* ap = lds_a + arow * KPAD + c * 32 + koff0;
        A.s.lo = *(const u32x4*)(ap);
        A.s.hi = *(const u32x4*)(ap + 16);
#pragma unroll
        for (int t = 0; t < 8; ++t) {
            BFfrag B;
            const unsigned int* bp = lds_w + (((t * kchunks + c) * 32 + lane) << 3);
            B.s.lo = *(const u32x4*)(bp);
            B.s.hi = *(const u32x4*)(bp + 4);
            acc[t] = __builtin_amdgcn_wmma_f32_16x16x32_bf16(
                false, A.v, false, B.v, (short)0, acc[t], false, false);
        }
    }
}

// bias + ReLU + LayerNorm, write bf16 activations back as next layer's A.
// C-tile layout: vgpr g -> row (wave*16 + g + (lane<16?0:8)), col = t*16+lane%16.
template <int KPAD>
__device__ __forceinline__ void ln_relu_store(v8f* acc,
                                              const float* __restrict__ bias,
                                              const float* __restrict__ gamma,
                                              const float* __restrict__ beta,
                                              unsigned short* __restrict__ lds_a,
                                              int wave, int lane) {
    const int col  = lane & 15;
    const int rsel = (lane < 16) ? 0 : 8;
    float bz[8], gz[8], bt[8];
#pragma unroll
    for (int t = 0; t < 8; ++t) {
        bz[t] = bias [t * 16 + col];
        gz[t] = gamma[t * 16 + col];
        bt[t] = beta [t * 16 + col];
    }
#pragma unroll
    for (int t = 0; t < 8; ++t)
#pragma unroll
        for (int g = 0; g < 8; ++g)
            acc[t][g] = fmaxf(acc[t][g] + bz[t], 0.f);

#pragma unroll
    for (int g = 0; g < 8; ++g) {
        float s = 0.f;
#pragma unroll
        for (int t = 0; t < 8; ++t) s += acc[t][g];
        for (int m = 1; m < 16; m <<= 1) s += __shfl_xor(s, m, 16);
        float mu = s * (1.0f / 128.0f);
        float sv = 0.f;
#pragma unroll
        for (int t = 0; t < 8; ++t) { float d = acc[t][g] - mu; sv += d * d; }
        for (int m = 1; m < 16; m <<= 1) sv += __shfl_xor(sv, m, 16);
        float rstd = rsqrtf(sv * (1.0f / 128.0f) + 1e-5f);
        int row = wave * 16 + g + rsel;
#pragma unroll
        for (int t = 0; t < 8; ++t) {
            float v = (acc[t][g] - mu) * rstd * gz[t] + bt[t];
            lds_a[row * KPAD + t * 16 + col] = f2bf(v);
        }
    }
}

__device__ __forceinline__ void zero_acc(v8f* acc) {
#pragma unroll
    for (int t = 0; t < 8; ++t)
#pragma unroll
        for (int g = 0; g < 8; ++g) acc[t][g] = 0.f;
}

// -------------------------------- edge MLP --------------------------------
// 128 edges per block, K0 = 260 padded to 288 (9 k-chunks), then 3x K=128.
// wfrag offsets (u32): L0=0 (18432), L1=18432, L2=26624, L3=34816.
#define KPAD_E 288

__global__ void __launch_bounds__(256)
edge_mlp_kernel(const float* __restrict__ x, const float* __restrict__ ea,
                const int* __restrict__ eidx,
                const float* __restrict__ b0, const float* __restrict__ g0, const float* __restrict__ be0,
                const float* __restrict__ b1, const float* __restrict__ g1, const float* __restrict__ be1,
                const float* __restrict__ b2, const float* __restrict__ g2, const float* __restrict__ be2,
                const float* __restrict__ b3,
                const unsigned int* __restrict__ wfrag,
                float* __restrict__ edge_out, float* __restrict__ agg) {
    extern __shared__ char smem[];
    unsigned int*   lds_w = (unsigned int*)smem;                 // up to 73728 B
    unsigned short* lds_a = (unsigned short*)(smem + 73728);     // 128 x 288 bf16

    const int tid  = threadIdx.x;
    const int lane = tid & 31;
    const int wave = tid >> 5;
    const int e0   = blockIdx.x * 128;

    // Stage layer-0 weight fragments + build the gathered A tile.
    copy_w(wfrag, lds_w, 18432, tid);
    for (int idx = tid; idx < 128 * KPAD_E; idx += 256) {
        int r = idx / KPAD_E, c = idx - r * KPAD_E;
        int e = e0 + r;
        float v = 0.f;
        if (c < 4)        v = ea[(size_t)e * 4 + c];
        else if (c < 132) v = x[(size_t)eidx[e] * HID + (c - 4)];
        else if (c < 260) v = x[(size_t)eidx[N_EDGES + e] * HID + (c - 132)];
        lds_a[idx] = f2bf(v);
    }
    __syncthreads();

    v8f acc[8];
    zero_acc(acc);
    wmma_gemm<KPAD_E>(lds_a, lds_w, 9, wave, lane, acc);
    ln_relu_store<KPAD_E>(acc, b0, g0, be0, lds_a, wave, lane);
    __syncthreads();

    copy_w(wfrag + 18432, lds_w, 8192, tid);
    __syncthreads();
    zero_acc(acc);
    wmma_gemm<KPAD_E>(lds_a, lds_w, 4, wave, lane, acc);
    ln_relu_store<KPAD_E>(acc, b1, g1, be1, lds_a, wave, lane);
    __syncthreads();

    copy_w(wfrag + 26624, lds_w, 8192, tid);
    __syncthreads();
    zero_acc(acc);
    wmma_gemm<KPAD_E>(lds_a, lds_w, 4, wave, lane, acc);
    ln_relu_store<KPAD_E>(acc, b2, g2, be2, lds_a, wave, lane);
    __syncthreads();

    copy_w(wfrag + 34816, lds_w, 8192, tid);
    __syncthreads();
    zero_acc(acc);
    wmma_gemm<KPAD_E>(lds_a, lds_w, 4, wave, lane, acc);

    // Final: edge_out store + segment-sum atomics into agg (= d_out head).
    const int col = lane & 15;
#pragma unroll
    for (int g = 0; g < 8; ++g) {
        int row = wave * 16 + g + ((lane < 16) ? 0 : 8);
        int e   = e0 + row;
        int rv  = eidx[N_EDGES + e];
#pragma unroll
        for (int t = 0; t < 8; ++t) {
            float v = acc[t][g] + b3[t * 16 + col];
            edge_out[(size_t)e * HID + t * 16 + col] = v;
            atomicAdd(&agg[(size_t)rv * HID + t * 16 + col], v);
        }
    }
}

// -------------------------------- node MLP --------------------------------
// 128 nodes per block (391 blocks, tail guarded), K0 = 256 (8 chunks).
// wfrag offsets (u32): L0=43008 (16384), L1=59392, L2=67584, L3=75776.
#define KPAD_N 256

__global__ void __launch_bounds__(256)
node_mlp_kernel(const float* __restrict__ x, const float* __restrict__ agg,
                const float* __restrict__ b0, const float* __restrict__ g0, const float* __restrict__ be0,
                const float* __restrict__ b1, const float* __restrict__ g1, const float* __restrict__ be1,
                const float* __restrict__ b2, const float* __restrict__ g2, const float* __restrict__ be2,
                const float* __restrict__ b3,
                const unsigned int* __restrict__ wfrag,
                float* __restrict__ out_x) {
    extern __shared__ char smem[];
    unsigned int*   lds_w = (unsigned int*)smem;                 // up to 65536 B
    unsigned short* lds_a = (unsigned short*)(smem + 65536);     // 128 x 256 bf16

    const int tid  = threadIdx.x;
    const int lane = tid & 31;
    const int wave = tid >> 5;
    const int n0   = blockIdx.x * 128;

    copy_w(wfrag + 43008, lds_w, 16384, tid);
    for (int idx = tid; idx < 128 * KPAD_N; idx += 256) {
        int r = idx >> 8, c = idx & 255;
        int n = n0 + r;
        float v = 0.f;
        if (n < N_NODES)
            v = (c < HID) ? x[(size_t)n * HID + c] : agg[(size_t)n * HID + (c - HID)];
        lds_a[idx] = f2bf(v);
    }
    __syncthreads();

    v8f acc[8];
    zero_acc(acc);
    wmma_gemm<KPAD_N>(lds_a, lds_w, 8, wave, lane, acc);
    ln_relu_store<KPAD_N>(acc, b0, g0, be0, lds_a, wave, lane);
    __syncthreads();

    copy_w(wfrag + 59392, lds_w, 8192, tid);
    __syncthreads();
    zero_acc(acc);
    wmma_gemm<KPAD_N>(lds_a, lds_w, 4, wave, lane, acc);
    ln_relu_store<KPAD_N>(acc, b1, g1, be1, lds_a, wave, lane);
    __syncthreads();

    copy_w(wfrag + 67584, lds_w, 8192, tid);
    __syncthreads();
    zero_acc(acc);
    wmma_gemm<KPAD_N>(lds_a, lds_w, 4, wave, lane, acc);
    ln_relu_store<KPAD_N>(acc, b2, g2, be2, lds_a, wave, lane);
    __syncthreads();

    copy_w(wfrag + 75776, lds_w, 8192, tid);
    __syncthreads();
    zero_acc(acc);
    wmma_gemm<KPAD_N>(lds_a, lds_w, 4, wave, lane, acc);

    const int col = lane & 15;
#pragma unroll
    for (int g = 0; g < 8; ++g) {
        int row = wave * 16 + g + ((lane < 16) ? 0 : 8);
        int n   = n0 + row;
        if (n < N_NODES) {
#pragma unroll
            for (int t = 0; t < 8; ++t) {
                float v = acc[t][g] + b3[t * 16 + col] + x[(size_t)n * HID + t * 16 + col];
                out_x[(size_t)n * HID + t * 16 + col] = v;   // overwrites agg row it owns
            }
        }
    }
}

// ------------------------------- launch glue -------------------------------
extern "C" void kernel_launch(void* const* d_in, const int* in_sizes, int n_in,
                              void* d_out, int out_size, void* d_ws, size_t ws_size,
                              hipStream_t stream) {
    const float* x    = (const float*)d_in[0];
    const float* ea   = (const float*)d_in[1];
    const int*   eidx = (const int*)d_in[2];

    // edge_params: W0 b0 g0 be0 W1 b1 g1 be1 W2 b2 g2 be2 W3 b3  -> d_in[3..16]
    const float* eW[4] = {(const float*)d_in[3], (const float*)d_in[7],
                          (const float*)d_in[11], (const float*)d_in[15]};
    const float* eb[4] = {(const float*)d_in[4], (const float*)d_in[8],
                          (const float*)d_in[12], (const float*)d_in[16]};
    const float* eg[3] = {(const float*)d_in[5], (const float*)d_in[9], (const float*)d_in[13]};
    const float* ez[3] = {(const float*)d_in[6], (const float*)d_in[10], (const float*)d_in[14]};
    // node_params -> d_in[17..30]
    const float* nW[4] = {(const float*)d_in[17], (const float*)d_in[21],
                          (const float*)d_in[25], (const float*)d_in[29]};
    const float* nb[4] = {(const float*)d_in[18], (const float*)d_in[22],
                          (const float*)d_in[26], (const float*)d_in[30]};
    const float* ng[3] = {(const float*)d_in[19], (const float*)d_in[23], (const float*)d_in[27]};
    const float* nz[3] = {(const float*)d_in[20], (const float*)d_in[24], (const float*)d_in[28]};

    unsigned int* wfrag  = (unsigned int*)d_ws;                  // ~336 KB used
    float* out_head      = (float*)d_out;                        // (x+x_new), also agg
    float* edge_out      = (float*)d_out + (size_t)N_NODES * HID;

    (void)hipFuncSetAttribute((const void*)edge_mlp_kernel,
                              hipFuncAttributeMaxDynamicSharedMemorySize, 147456);
    (void)hipFuncSetAttribute((const void*)node_mlp_kernel,
                              hipFuncAttributeMaxDynamicSharedMemorySize, 131072);

    // Pre-swizzle weights into WMMA B-fragment layout (bf16).
    prep_w_kernel<<<8 * 9, 256, 0, stream>>>(eW[0], 260, 9, wfrag + 0);
    prep_w_kernel<<<8 * 4, 256, 0, stream>>>(eW[1], 128, 4, wfrag + 18432);
    prep_w_kernel<<<8 * 4, 256, 0, stream>>>(eW[2], 128, 4, wfrag + 26624);
    prep_w_kernel<<<8 * 4, 256, 0, stream>>>(eW[3], 128, 4, wfrag + 34816);
    prep_w_kernel<<<8 * 8, 256, 0, stream>>>(nW[0], 256, 8, wfrag + 43008);
    prep_w_kernel<<<8 * 4, 256, 0, stream>>>(nW[1], 128, 4, wfrag + 59392);
    prep_w_kernel<<<8 * 4, 256, 0, stream>>>(nW[2], 128, 4, wfrag + 67584);
    prep_w_kernel<<<8 * 4, 256, 0, stream>>>(nW[3], 128, 4, wfrag + 75776);

    // Zero the agg region (head of d_out) every call — we accumulate into it.
    zero_agg_kernel<<<6250, 256, 0, stream>>>(out_head);

    edge_mlp_kernel<<<N_EDGES / 128, 256, 147456, stream>>>(
        x, ea, eidx,
        eb[0], eg[0], ez[0], eb[1], eg[1], ez[1], eb[2], eg[2], ez[2], eb[3],
        wfrag, edge_out, out_head);

    node_mlp_kernel<<<(N_NODES + 127) / 128, 256, 131072, stream>>>(
        x, out_head,
        nb[0], ng[0], nz[0], nb[1], ng[1], nz[1], nb[2], ng[2], nz[2], nb[3],
        wfrag, out_head);
}
